// HGAIT_19207093747767
// MI455X (gfx1250) — matrix operand order repaired and checked by reading, not compiled, not run on hardware
//
#include <hip/hip_runtime.h>
#include <math.h>

// MI455X / gfx1250, wave32. All dense GEMMs via v_wmma_f32_16x16x32_f16.
// GEMM requires Nd % 64 == 0 and Kd % 64 == 0; the 6000-wide similarity
// GEMM is padded to 6016 columns.

typedef __attribute__((ext_vector_type(16))) _Float16 v16h;
typedef __attribute__((ext_vector_type(8)))  float    v8f;

constexpr int Nn   = 6000;
constexpr int NSIM = 6016;               // 6000 padded to multiple of 64
constexpr int Ff   = 16;
constexpr int Ss   = 64;
constexpr int Dd   = 256;
constexpr int Hh   = 8;
constexpr int Kk   = 32;
constexpr int DHh  = 32;
constexpr int Ee   = Nn*Kk + Nn;         // 198000 edges (incl. self loops)

__device__ __forceinline__ float gelu_f(float x){
  return 0.5f * x * (1.f + erff(x * 0.70710678118654752f));
}
__device__ __forceinline__ float sigmoid_f(float x){ return 1.f/(1.f+expf(-x)); }

__device__ __forceinline__ float blockReduceSum(float v, float* red){
  int tid = threadIdx.x;
  red[tid] = v; __syncthreads();
  for (int s = blockDim.x>>1; s>0; s>>=1){
    if (tid < s) red[tid] += red[tid+s];
    __syncthreads();
  }
  float r = red[0]; __syncthreads();
  return r;
}

__device__ __forceinline__ void atomicMaxFloat(float* addr, float val){
  if (val >= 0.f) atomicMax((int*)addr, __float_as_int(val));
  else            atomicMin((unsigned int*)addr, __float_as_uint(val));
}

__device__ __forceinline__ v16h cvt_a(float4 f0, float4 f1, float4 f2, float4 f3){
  v16h a;
  a[0]=(_Float16)f0.x;  a[1]=(_Float16)f0.y;  a[2]=(_Float16)f0.z;  a[3]=(_Float16)f0.w;
  a[4]=(_Float16)f1.x;  a[5]=(_Float16)f1.y;  a[6]=(_Float16)f1.z;  a[7]=(_Float16)f1.w;
  a[8]=(_Float16)f2.x;  a[9]=(_Float16)f2.y;  a[10]=(_Float16)f2.z; a[11]=(_Float16)f2.w;
  a[12]=(_Float16)f3.x; a[13]=(_Float16)f3.y; a[14]=(_Float16)f3.z; a[15]=(_Float16)f3.w;
  return a;
}

// ---------------------------------------------------------------------------
// Generic WMMA GEMM:  C[b] = A[b] (MxK, f32, row stride lda) @ B (KxN) + bias
// Bt[n*K + k] (f16) -> contiguous 32B per-lane B-fragment loads (k = 16*half + s,
// same order as the A loads). Each wave computes one 16x64 strip.
// All loads of a double K-step are hoisted ahead of conversion/WMMA so the
// compiler issues them as one clause with a single s_wait_loadcnt.
// REQUIREMENT: Nd % 64 == 0, M % 16 == 0, Kd % 64 == 0.
// ---------------------------------------------------------------------------
__global__ __launch_bounds__(256)
void k_gemm(const float* __restrict__ A, long long lda, long long sA,
            const _Float16* __restrict__ Bt, long long sB,
            const float* __restrict__ bias, long long sbias, int has_bias,
            float* __restrict__ C, long long ldc, long long sC,
            int M, int Kd, int Nd)
{
  int b = blockIdx.y;
  A  += (long long)b * sA;
  Bt += (long long)b * sB;
  C  += (long long)b * sC;
  if (has_bias) bias += (long long)b * sbias;

  int lane = threadIdx.x & 31;
  int wid  = blockIdx.x * (blockDim.x >> 5) + (threadIdx.x >> 5);
  int tilesN = Nd >> 6;
  int mt = wid / tilesN;
  int nb = wid - mt * tilesN;
  if (mt * 16 >= M) return;
  int m0 = mt * 16, n0 = nb * 64;
  int arow = m0 + (lane & 15);
  int hf   = lane >> 4;                  // which half of the wave

  v8f acc0, acc1, acc2, acc3;
  #pragma unroll
  for (int e = 0; e < 8; e++){ acc0[e]=0.f; acc1[e]=0.f; acc2[e]=0.f; acc3[e]=0.f; }

  const _Float16* bp0 = Bt + (size_t)(n0      + (lane & 15)) * Kd + hf * 16;
  const _Float16* bp1 = Bt + (size_t)(n0 + 16 + (lane & 15)) * Kd + hf * 16;
  const _Float16* bp2 = Bt + (size_t)(n0 + 32 + (lane & 15)) * Kd + hf * 16;
  const _Float16* bp3 = Bt + (size_t)(n0 + 48 + (lane & 15)) * Kd + hf * 16;
  const float*    ap  = A  + (size_t)arow * lda + hf * 16;

  for (int kk = 0; kk < Kd; kk += 64){
    // ---- issue ALL loads for two K-steps before any use ----
    const float4* av0 = (const float4*)(ap + kk);        // K-step kk
    const float4* av1 = (const float4*)(ap + kk + 32);   // K-step kk+32
    float4 fa0 = av0[0], fa1 = av0[1], fa2 = av0[2], fa3 = av0[3];
    float4 ga0 = av1[0], ga1 = av1[1], ga2 = av1[2], ga3 = av1[3];
    v16h b00 = *(const v16h*)(bp0 + kk);
    v16h b01 = *(const v16h*)(bp1 + kk);
    v16h b02 = *(const v16h*)(bp2 + kk);
    v16h b03 = *(const v16h*)(bp3 + kk);
    v16h b10 = *(const v16h*)(bp0 + kk + 32);
    v16h b11 = *(const v16h*)(bp1 + kk + 32);
    v16h b12 = *(const v16h*)(bp2 + kk + 32);
    v16h b13 = *(const v16h*)(bp3 + kk + 32);

    // ---- convert + compute ----
    v16h a0 = cvt_a(fa0, fa1, fa2, fa3);
    acc0 = __builtin_amdgcn_wmma_f32_16x16x32_f16(false, a0, false, b00, (short)0, acc0, false, false);
    acc1 = __builtin_amdgcn_wmma_f32_16x16x32_f16(false, a0, false, b01, (short)0, acc1, false, false);
    acc2 = __builtin_amdgcn_wmma_f32_16x16x32_f16(false, a0, false, b02, (short)0, acc2, false, false);
    acc3 = __builtin_amdgcn_wmma_f32_16x16x32_f16(false, a0, false, b03, (short)0, acc3, false, false);
    v16h a1 = cvt_a(ga0, ga1, ga2, ga3);
    acc0 = __builtin_amdgcn_wmma_f32_16x16x32_f16(false, a1, false, b10, (short)0, acc0, false, false);
    acc1 = __builtin_amdgcn_wmma_f32_16x16x32_f16(false, a1, false, b11, (short)0, acc1, false, false);
    acc2 = __builtin_amdgcn_wmma_f32_16x16x32_f16(false, a1, false, b12, (short)0, acc2, false, false);
    acc3 = __builtin_amdgcn_wmma_f32_16x16x32_f16(false, a1, false, b13, (short)0, acc3, false, false);
  }

  // C/D layout: element e -> row m0 + hf*8 + e, col n0 + j*16 + (lane&15)
  int mbase = m0 + hf * 8;
  int ncol  = lane & 15;
  float bb0 = has_bias ? bias[n0      + ncol] : 0.f;
  float bb1 = has_bias ? bias[n0 + 16 + ncol] : 0.f;
  float bb2 = has_bias ? bias[n0 + 32 + ncol] : 0.f;
  float bb3 = has_bias ? bias[n0 + 48 + ncol] : 0.f;
  #pragma unroll
  for (int e = 0; e < 8; e++){
    float* crow = C + (size_t)(mbase + e) * ldc + n0 + ncol;
    crow[0]  = acc0[e] + bb0;
    crow[16] = acc1[e] + bb1;
    crow[32] = acc2[e] + bb2;
    crow[48] = acc3[e] + bb3;
  }
}

// ------------------------- weight prep -------------------------------------
__global__ void k_transpose_f16(const float* __restrict__ in, _Float16* __restrict__ out,
                                int Kd, int Nd){
  int t = blockIdx.x * blockDim.x + threadIdx.x;
  if (t >= Kd * Nd) return;
  int k = t / Nd, n = t - k * Nd;
  out[(size_t)n * Kd + k] = (_Float16)in[t];
}
__global__ void k_f32_to_f16(const float* __restrict__ in, _Float16* __restrict__ out, int n){
  int t = blockIdx.x * blockDim.x + threadIdx.x;
  if (t < n) out[t] = (_Float16)in[t];
}
__global__ void k_fill(float* __restrict__ p, float v, int n){
  int t = blockIdx.x * blockDim.x + threadIdx.x;
  if (t < n) p[t] = v;
}
__global__ void k_fill_h(_Float16* __restrict__ p, int n){
  int t = blockIdx.x * blockDim.x + threadIdx.x;
  if (t < n) p[t] = (_Float16)0.f;
}

// ------------------------- GRU gates ---------------------------------------
__global__ void k_gru_gate(const float* __restrict__ g, const float* __restrict__ bhh,
                           float* __restrict__ emb){
  int t = blockIdx.x * blockDim.x + threadIdx.x;
  if (t >= Nn * Ff * Dd) return;
  int d = t & 255;
  int f = (t >> 8) & 15;
  size_t row = (size_t)(t >> 12) * Ff + f;
  size_t gb = row * 768;
  float gr = g[gb + d], gz = g[gb + 256 + d], gn = g[gb + 512 + d];
  float br = bhh[f*768 + d], bz = bhh[f*768 + 256 + d], bn = bhh[f*768 + 512 + d];
  float r = sigmoid_f(gr + br);
  float z = sigmoid_f(gz + bz);
  emb[row * 256 + d] = (1.f - z) * tanhf(gn + r * bn);
}

// ------------------------- LayerNorm (rows of 256) --------------------------
// mode 0: y = LN(x) ; mode 1: y = LN(gelu(x) + res)
__global__ __launch_bounds__(256)
void k_ln(const float* __restrict__ x, const float* __restrict__ res,
          const float* __restrict__ g, const float* __restrict__ b,
          float* __restrict__ y, int mode){
  __shared__ float red[256];
  int tid = threadIdx.x;
  size_t base = (size_t)blockIdx.x * 256 + tid;
  float v = x[base];
  if (mode == 1) v = gelu_f(v) + res[base];
  float m   = blockReduceSum(v, red) * (1.f/256.f);
  float dv  = v - m;
  float var = blockReduceSum(dv*dv, red) * (1.f/256.f);
  y[base] = dv * rsqrtf(var + 1e-5f) * g[tid] + b[tid];
}

// ------------------------- per-node attention over F=16 tokens --------------
__global__ __launch_bounds__(128)
void k_attn(const float* __restrict__ q, const float* __restrict__ k,
            const float* __restrict__ v, float* __restrict__ o){
  __shared__ float sq[Ff*Dd], sk[Ff*Dd], sv[Ff*Dd];
  size_t base = (size_t)blockIdx.x * Ff * Dd;
  for (int i = threadIdx.x; i < Ff*Dd; i += 128){
    sq[i] = q[base + i]; sk[i] = k[base + i]; sv[i] = v[base + i];
  }
  __syncthreads();
  int t = threadIdx.x;
  int h = t >> 4, c = t & 15;
  const float scale = 0.17677669529663687f;    // 1/sqrt(32)
  float row[16]; float mx = -3e38f;
  #pragma unroll
  for (int s = 0; s < 16; s++){
    float acc = 0.f;
    #pragma unroll
    for (int i = 0; i < DHh; i++) acc += sq[c*Dd + h*DHh + i] * sk[s*Dd + h*DHh + i];
    acc *= scale; row[s] = acc; mx = fmaxf(mx, acc);
  }
  float sum = 0.f;
  #pragma unroll
  for (int s = 0; s < 16; s++){ row[s] = expf(row[s] - mx); sum += row[s]; }
  float inv = 1.f / sum;
  for (int i = 0; i < DHh; i++){
    float acc = 0.f;
    #pragma unroll
    for (int s = 0; s < 16; s++) acc += row[s] * sv[s*Dd + h*DHh + i];
    o[base + c*Dd + h*DHh + i] = acc * inv;
  }
}

// ------------------------- feature-importance softmax -----------------------
__global__ __launch_bounds__(256)
void k_fi(const float* __restrict__ u, const float* __restrict__ W2,
          const float* __restrict__ b2, float* __restrict__ fi){
  __shared__ float red[256];
  __shared__ float wv[16];
  int n = blockIdx.x, tid = threadIdx.x;
  float w2 = W2[tid];
  for (int f = 0; f < 16; f++){
    float x = u[((size_t)n * 16 + f) * 256 + tid];
    float s = blockReduceSum(gelu_f(x) * w2, red);
    if (tid == 0) wv[f] = s + b2[0];
  }
  __syncthreads();
  if (tid < 16){
    float mx = -3e38f;
    for (int f = 0; f < 16; f++) mx = fmaxf(mx, wv[f]);
    float sum = 0.f;
    for (int f = 0; f < 16; f++) sum += expf(wv[f] - mx);
    fi[n*16 + tid] = expf(wv[tid] - mx) / sum;
  }
}

__global__ void k_node(const float* __restrict__ fi, const float* __restrict__ emb,
                       float* __restrict__ node){
  int t = blockIdx.x * blockDim.x + threadIdx.x;
  if (t >= Nn * Dd) return;
  int n = t >> 8, d = t & 255;
  float acc = 0.f;
  #pragma unroll
  for (int f = 0; f < 16; f++)
    acc += fi[n*16 + f] * emb[((size_t)n*16 + f) * 256 + d];
  node[t] = acc;
}

__global__ __launch_bounds__(256)
void k_norm(const float* __restrict__ node, float* __restrict__ xn, _Float16* __restrict__ xnH){
  __shared__ float red[256];
  int tid = threadIdx.x;
  size_t base = (size_t)blockIdx.x * 256 + tid;
  float v = node[base];
  float ss = blockReduceSum(v*v, red);
  float o = v * rsqrtf(ss + 1e-12f);
  xn[base] = o; xnH[base] = (_Float16)o;
}

// ------------------------- top/bottom-K per row ------------------------------
__global__ __launch_bounds__(256)
void k_topk(const float* __restrict__ sim, int* __restrict__ out_idx, int flip){
  __shared__ float ls[Nn];
  __shared__ float rv[256];
  __shared__ int   ri[256];
  int n = blockIdx.x, tid = threadIdx.x;
  const float* row = sim + (size_t)n * NSIM;     // padded leading dim
  for (int i = tid; i < Nn; i += 256){ float v = row[i]; ls[i] = flip ? -v : v; }
  __syncthreads();
  if (tid == 0) ls[n] = -3e38f;          // exclude self
  __syncthreads();
  for (int it = 0; it < Kk; it++){
    float bv = -3e38f; int bi = 0;
    for (int i = tid; i < Nn; i += 256) if (ls[i] > bv){ bv = ls[i]; bi = i; }
    rv[tid] = bv; ri[tid] = bi; __syncthreads();
    for (int s = 128; s > 0; s >>= 1){
      if (tid < s && rv[tid + s] > rv[tid]){ rv[tid] = rv[tid + s]; ri[tid] = ri[tid + s]; }
      __syncthreads();
    }
    if (tid == 0){ out_idx[n*Kk + it] = ri[0]; ls[ri[0]] = -3e38f; }
    __syncthreads();
  }
}

// ------------------------- GAT pieces ---------------------------------------
__global__ void k_esed(const float* __restrict__ h1, const float* __restrict__ asrc,
                       const float* __restrict__ adst, float* __restrict__ es,
                       float* __restrict__ ed){
  int t = blockIdx.x * blockDim.x + threadIdx.x;
  if (t >= Nn * Hh) return;
  int n = t / Hh, h = t - n * Hh;
  float s = 0.f, d = 0.f;
  #pragma unroll
  for (int i = 0; i < DHh; i++){
    float hv = h1[(size_t)n*256 + h*DHh + i];
    s += hv * asrc[h*DHh + i];
    d += hv * adst[h*DHh + i];
  }
  es[t] = s; ed[t] = d;
}

// pass 0: segment max (atomic); pass 1: exp-sum; pass 2: weighted aggregation
__global__ void k_edge(const int* __restrict__ idx, const float* __restrict__ es,
                       const float* __restrict__ ed, float* __restrict__ mseg,
                       float* __restrict__ sseg, const float* __restrict__ h1,
                       float* __restrict__ agg, int pass){
  int t = blockIdx.x * blockDim.x + threadIdx.x;
  if (t >= Ee * Hh) return;
  int e = t / Hh, h = t - e * Hh;
  int src, tgt; bool valid;
  if (e < Nn * Kk){ src = e / Kk; tgt = idx[e]; valid = (src != tgt); }
  else            { src = tgt = e - Nn * Kk; valid = true; }
  if (!valid) return;
  float ev = es[src*Hh + h] + ed[tgt*Hh + h];
  ev = ev > 0.f ? ev : 0.2f * ev;                         // leaky_relu 0.2
  if (pass == 0){
    atomicMaxFloat(&mseg[tgt*Hh + h], ev);
  } else if (pass == 1){
    atomicAdd(&sseg[tgt*Hh + h], expf(ev - mseg[tgt*Hh + h]));
  } else {
    float alpha = expf(ev - mseg[tgt*Hh + h]) / (sseg[tgt*Hh + h] + 1e-16f);
    #pragma unroll
    for (int i = 0; i < DHh; i++)
      atomicAdd(&agg[(size_t)tgt*256 + h*DHh + i], alpha * h1[(size_t)src*256 + h*DHh + i]);
  }
}

// ------------------------- importance fuse + double LN ----------------------
__global__ __launch_bounds__(256)
void k_fuse(const float* __restrict__ selfO, const float* __restrict__ aggT,
            const float* __restrict__ aggB, const float* __restrict__ gatBias,
            const float* __restrict__ impW, const float* __restrict__ impB,
            const float* __restrict__ node, const float* __restrict__ gaG,
            const float* __restrict__ gaB, const float* __restrict__ fnG,
            const float* __restrict__ fnB, float* __restrict__ out2){
  __shared__ float red[256];
  int tid = threadIdx.x;
  size_t base = (size_t)blockIdx.x * 256 + tid;
  float a0 = selfO[base];
  float a1 = aggT[base] + gatBias[256 + tid];
  float a2 = aggB[base] + gatBias[512 + tid];
  float l0 = blockReduceSum(a0 * impW[tid],       red) + impB[0];
  float l1 = blockReduceSum(a1 * impW[256 + tid], red) + impB[1];
  float l2 = blockReduceSum(a2 * impW[512 + tid], red) + impB[2];
  float mx = fmaxf(l0, fmaxf(l1, l2));
  float e0 = expf(l0 - mx), e1 = expf(l1 - mx), e2 = expf(l2 - mx);
  float inv = 1.f / (e0 + e1 + e2);
  float fused = (e0*a0 + e1*a1 + e2*a2) * inv;
  float vv = gelu_f(fused) + node[base];
  float m1 = blockReduceSum(vv, red) * (1.f/256.f);
  float d1 = vv - m1;
  float var1 = blockReduceSum(d1*d1, red) * (1.f/256.f);
  float y = d1 * rsqrtf(var1 + 1e-5f) * gaG[tid] + gaB[tid];
  float m2 = blockReduceSum(y, red) * (1.f/256.f);
  float d2 = y - m2;
  float var2 = blockReduceSum(d2*d2, red) * (1.f/256.f);
  out2[base] = d2 * rsqrtf(var2 + 1e-5f) * fnG[tid] + fnB[tid];
}

// ------------------------- predictor reduction ------------------------------
__global__ __launch_bounds__(256)
void k_pred(const float* __restrict__ p1, const float* __restrict__ W2,
            const float* __restrict__ b2, float* __restrict__ out){
  __shared__ float red[256];
  int n = blockIdx.x, tid = threadIdx.x;
  float acc = gelu_f(p1[(size_t)n*512 + tid])       * W2[tid]
            + gelu_f(p1[(size_t)n*512 + 256 + tid]) * W2[256 + tid];
  float s = blockReduceSum(acc, red);
  if (tid == 0) out[n] = s + b2[0];
}

// ===========================================================================
extern "C" void kernel_launch(void* const* d_in, const int* in_sizes, int n_in,
                              void* d_out, int out_size, void* d_ws, size_t ws_size,
                              hipStream_t stream)
{
  (void)in_sizes; (void)n_in; (void)out_size; (void)ws_size;
  const float* x      = (const float*)d_in[0];
  const float* gruWih = (const float*)d_in[1];
  const float* gruBih = (const float*)d_in[2];
  const float* gruBhh = (const float*)d_in[3];
  const float* vaWq   = (const float*)d_in[4];
  const float* vaBq   = (const float*)d_in[5];
  const float* vaWk   = (const float*)d_in[6];
  const float* vaBk   = (const float*)d_in[7];
  const float* vaWv   = (const float*)d_in[8];
  const float* vaBv   = (const float*)d_in[9];
  const float* lnqG   = (const float*)d_in[10];
  const float* lnqB   = (const float*)d_in[11];
  const float* lnkG   = (const float*)d_in[12];
  const float* lnkB   = (const float*)d_in[13];
  const float* lnvG   = (const float*)d_in[14];
  const float* lnvB   = (const float*)d_in[15];
  const float* vaWo   = (const float*)d_in[16];
  const float* vaBo   = (const float*)d_in[17];
  const float* vaLnG  = (const float*)d_in[18];
  const float* vaLnB  = (const float*)d_in[19];
  const float* vaFi1  = (const float*)d_in[20];
  const float* vaFib1 = (const float*)d_in[21];
  const float* vaFiW2 = (const float*)d_in[22];
  const float* vaFib2 = (const float*)d_in[23];
  const float* gatW   = (const float*)d_in[24];
  const float* gatAs  = (const float*)d_in[25];
  const float* gatAd  = (const float*)d_in[26];
  const float* gatBias= (const float*)d_in[27];
  const float* impW   = (const float*)d_in[28];
  const float* impB   = (const float*)d_in[29];
  const float* gaG    = (const float*)d_in[30];
  const float* gaB    = (const float*)d_in[31];
  const float* fnG    = (const float*)d_in[32];
  const float* fnB    = (const float*)d_in[33];
  const float* predW1 = (const float*)d_in[34];
  const float* predB1 = (const float*)d_in[35];
  const float* predW2 = (const float*)d_in[36];
  const float* predB2 = (const float*)d_in[37];
  float* out = (float*)d_out;

  char* base = (char*)d_ws;
  size_t off = 0;
  auto alloc = [&](size_t bytes) -> void* {
    off = (off + 255) & ~(size_t)255;
    void* p = base + off;
    off += bytes;
    return p;
  };

  const size_t NF = (size_t)Nn * Ff;                  // 96000
  _Float16* hProj = (_Float16*)alloc(10u * 65536 * 2);   // [l*5+{Wq,Wk,Wv,Wo,Fi1}]
  _Float16* hGat  = (_Float16*)alloc(3u  * 65536 * 2);
  _Float16* hPred = (_Float16*)alloc((size_t)512 * 256 * 2);
  _Float16* hGru  = (_Float16*)alloc((size_t)Ff * 768 * 64 * 2);
  _Float16* xnH   = (_Float16*)alloc((size_t)NSIM * 256 * 2); // 16 zero pad rows
  float* big  = (float*)alloc((size_t)Nn * Ff * 768 * 4);   // g -> q/k/v -> sim
  float* emb  = (float*)alloc(NF * 256 * 4);
  float* tA   = (float*)alloc(NF * 256 * 4);                // o, u, later hT/hB
  float* tB   = (float*)alloc(NF * 256 * 4);                // t, later selfO/agg/out2/p1
  float* fi   = (float*)alloc(NF * 4);
  float* node = (float*)alloc((size_t)Nn * 256 * 4);
  float* xn   = (float*)alloc((size_t)Nn * 256 * 4);
  float* es   = (float*)alloc((size_t)Nn * Hh * 4);
  float* ed   = (float*)alloc((size_t)Nn * Hh * 4);
  float* mseg = (float*)alloc((size_t)Nn * Hh * 4);
  float* sseg = (float*)alloc((size_t)Nn * Hh * 4);
  int*   idxT = (int*)alloc((size_t)Nn * Kk * 4);
  int*   idxB = (int*)alloc((size_t)Nn * Kk * 4);

  float* q  = big;
  float* km = big + NF * 256;
  float* v  = big + 2 * NF * 256;
  float* sim = big;                                   // Nn x NSIM (padded)
  float* hT    = tA;
  float* hB    = tA + (size_t)Nn * 256;
  float* selfO = tB;
  float* aggT  = tB + (size_t)Nn * 256;
  float* aggB  = tB + (size_t)2 * Nn * 256;
  float* out2  = tB + (size_t)3 * Nn * 256;
  float* p1    = tB + (size_t)4 * Nn * 256;           // Nn*512

  auto gemm = [&](const float* A, long long lda, long long sA,
                  const _Float16* Bt, long long sB,
                  const float* bias, long long sbias,
                  float* C, long long ldc, long long sC,
                  int M, int Kd, int Nd, int batch){
    int tilesN = Nd / 64;
    int waves  = (M / 16) * tilesN;
    dim3 grid((waves + 7) / 8, batch);
    k_gemm<<<grid, 256, 0, stream>>>(A, lda, sA, Bt, sB, bias, sbias,
                                     bias != nullptr, C, ldc, sC, M, Kd, Nd);
  };

  // ---- weight prep: f16 (transposed for row-major weights) ----
  const float* projSrc[10] = {
    vaWq, vaWk, vaWv, vaWo, vaFi1,
    vaWq + 65536, vaWk + 65536, vaWv + 65536, vaWo + 65536, vaFi1 + 65536 };
  for (int i = 0; i < 10; i++)
    k_transpose_f16<<<(65536 + 255)/256, 256, 0, stream>>>(projSrc[i], hProj + (size_t)i*65536, 256, 256);
  for (int g = 0; g < 3; g++)
    k_transpose_f16<<<(65536 + 255)/256, 256, 0, stream>>>(gatW + (size_t)g*65536, hGat + (size_t)g*65536, 256, 256);
  k_transpose_f16<<<(256*512 + 255)/256, 256, 0, stream>>>(predW1, hPred, 256, 512);
  // gru_Wih[f][k][s] is already Bt layout (N=3D rows of K=S) per feature
  k_f32_to_f16<<<(Ff*768*64 + 255)/256, 256, 0, stream>>>(gruWih, hGru, Ff*768*64);
  // zero the 16 pad rows of xn_h (cols 6000..6015 of sim become 0)
  k_fill_h<<<((NSIM-Nn)*256 + 255)/256, 256, 0, stream>>>(xnH + (size_t)Nn*256, (NSIM-Nn)*256);

  // ---- GRU time mixing: g[n,f,:] = x[n,f,:] @ Wih[f]^T + bih[f] ----
  gemm(x, (long long)Ff*Ss, Ss, hGru, (long long)768*64,
       gruBih, 768, big, (long long)Ff*768, 768, Nn, Ss, 768, Ff);
  k_gru_gate<<<(int)((NF*256 + 255)/256), 256, 0, stream>>>(big, gruBhh, emb);

  // ---- variable attention layers ----
  for (int l = 0; l < 2; l++){
    const _Float16* wq = hProj + (size_t)(l*5 + 0)*65536;
    const _Float16* wk = hProj + (size_t)(l*5 + 1)*65536;
    const _Float16* wv = hProj + (size_t)(l*5 + 2)*65536;
    const _Float16* wo = hProj + (size_t)(l*5 + 3)*65536;
    gemm(emb, 256, 0, wq, 0, vaBq + l*256, 0, q,  256, 0, (int)NF, 256, 256, 1);
    gemm(emb, 256, 0, wk, 0, vaBk + l*256, 0, km, 256, 0, (int)NF, 256, 256, 1);
    gemm(emb, 256, 0, wv, 0, vaBv + l*256, 0, v,  256, 0, (int)NF, 256, 256, 1);
    k_ln<<<(int)NF, 256, 0, stream>>>(q,  nullptr, lnqG + l*256, lnqB + l*256, q,  0);
    k_ln<<<(int)NF, 256, 0, stream>>>(km, nullptr, lnkG + l*256, lnkB + l*256, km, 0);
    k_ln<<<(int)NF, 256, 0, stream>>>(v,  nullptr, lnvG + l*256, lnvB + l*256, v,  0);
    k_attn<<<Nn, 128, 0, stream>>>(q, km, v, tA);
    gemm(tA, 256, 0, wo, 0, vaBo + l*256, 0, tB, 256, 0, (int)NF, 256, 256, 1);
    k_ln<<<(int)NF, 256, 0, stream>>>(tB, emb, vaLnG + l*256, vaLnB + l*256, emb, 1);
  }
  // feature-importance (only the last layer's fi is used)
  gemm(emb, 256, 0, hProj + (size_t)(1*5 + 4)*65536, 0, vaFib1 + 256, 0,
       tA, 256, 0, (int)NF, 256, 256, 1);
  k_fi<<<Nn, 256, 0, stream>>>(tA, vaFiW2 + 256, vaFib2 + 1, fi);
  k_node<<<(Nn*256 + 255)/256, 256, 0, stream>>>(fi, emb, node);

  // ---- similarity + top/bottom-K ----
  k_norm<<<Nn, 256, 0, stream>>>(node, xn, xnH);
  gemm(xn, 256, 0, xnH, 0, nullptr, 0, sim, NSIM, 0, Nn, 256, NSIM, 1);
  k_topk<<<Nn, 256, 0, stream>>>(sim, idxT, 0);
  k_topk<<<Nn, 256, 0, stream>>>(sim, idxB, 1);

  // ---- GAT: self branch ----
  gemm(node, 256, 0, hGat, 0, gatBias, 0, selfO, 256, 0, Nn, 256, 256, 1);

  const int EH = Ee * Hh;
  // top graph
  gemm(node, 256, 0, hGat + 65536, 0, nullptr, 0, hT, 256, 0, Nn, 256, 256, 1);
  k_esed<<<(Nn*Hh + 255)/256, 256, 0, stream>>>(hT, gatAs + 1*Hh*DHh, gatAd + 1*Hh*DHh, es, ed);
  k_fill<<<(Nn*Hh + 255)/256, 256, 0, stream>>>(mseg, -3e38f, Nn*Hh);
  k_fill<<<(Nn*Hh + 255)/256, 256, 0, stream>>>(sseg, 0.f, Nn*Hh);
  k_fill<<<(Nn*256 + 255)/256, 256, 0, stream>>>(aggT, 0.f, Nn*256);
  k_edge<<<(EH + 255)/256, 256, 0, stream>>>(idxT, es, ed, mseg, sseg, hT, aggT, 0);
  k_edge<<<(EH + 255)/256, 256, 0, stream>>>(idxT, es, ed, mseg, sseg, hT, aggT, 1);
  k_edge<<<(EH + 255)/256, 256, 0, stream>>>(idxT, es, ed, mseg, sseg, hT, aggT, 2);
  // bottom graph
  gemm(node, 256, 0, hGat + 2*65536, 0, nullptr, 0, hB, 256, 0, Nn, 256, 256, 1);
  k_esed<<<(Nn*Hh + 255)/256, 256, 0, stream>>>(hB, gatAs + 2*Hh*DHh, gatAd + 2*Hh*DHh, es, ed);
  k_fill<<<(Nn*Hh + 255)/256, 256, 0, stream>>>(mseg, -3e38f, Nn*Hh);
  k_fill<<<(Nn*Hh + 255)/256, 256, 0, stream>>>(sseg, 0.f, Nn*Hh);
  k_fill<<<(Nn*256 + 255)/256, 256, 0, stream>>>(aggB, 0.f, Nn*256);
  k_edge<<<(EH + 255)/256, 256, 0, stream>>>(idxB, es, ed, mseg, sseg, hB, aggB, 0);
  k_edge<<<(EH + 255)/256, 256, 0, stream>>>(idxB, es, ed, mseg, sseg, hB, aggB, 1);
  k_edge<<<(EH + 255)/256, 256, 0, stream>>>(idxB, es, ed, mseg, sseg, hB, aggB, 2);

  // ---- fuse + double LN + predictor ----
  k_fuse<<<Nn, 256, 0, stream>>>(selfO, aggT, aggB, gatBias, impW, impB,
                                 node, gaG, gaB, fnG, fnB, out2);
  gemm(out2, 256, 0, hPred, 0, predB1, 0, p1, 512, 0, Nn, 256, 512, 1);
  k_pred<<<Nn, 256, 0, stream>>>(p1, predW2, predB2, out);
}
